// DynamicConv2d_29609504539351
// MI455X (gfx1250) — matrix operand order repaired and checked by reading, not compiled, not run on hardware
//
#include <hip/hip_runtime.h>

typedef __attribute__((ext_vector_type(16))) _Float16 v16h;
typedef __attribute__((ext_vector_type(8)))  _Float16 v8h;
typedef __attribute__((ext_vector_type(8)))  float    v8f;

constexpr int IC  = 256;
constexpr int OC  = 512;
constexpr int HW  = 56;
constexpr int PW  = HW + 2;       // 58 padded width
constexpr int PP  = PW * PW;      // 3364 padded plane
constexpr int P   = HW * HW;      // 3136 spatial positions
constexpr int KD  = IC * 9;       // 2304 flattened filter dim
constexpr int NT  = 10;           // hash tables
constexpr int HS  = 8;            // hash bits
constexpr int LIM = OC / 2;       // top-k size
constexpr float EPS = 1e-3f;

// ---------------- precision conversion ----------------

// x f32 [IC][56][56] -> xp f16 [IC][58][58] with zero halo (kills all bounds checks in conv)
__global__ void dc_pad_x(const float* __restrict__ x, _Float16* __restrict__ xp) {
    int i = blockIdx.x * 256 + threadIdx.x;
    if (i >= IC * PP) return;
    int c  = i / PP;
    int r  = i - c * PP;
    int hh = r / PW;
    int ww = r - hh * PW;
    float v = 0.f;
    if (hh >= 1 && hh <= HW && ww >= 1 && ww <= HW)
        v = x[c * P + (hh - 1) * HW + (ww - 1)];
    xp[i] = (_Float16)v;
}

// w [OC][IC][3][3] f32  ->  wt [tap][OC][IC] f16   (makes K contiguous per filter)
__global__ void dc_cvt_w(const float* __restrict__ w, _Float16* __restrict__ wt) {
    int i = blockIdx.x * 256 + threadIdx.x;
    if (i >= 9 * OC * IC) return;
    int t  = i / (OC * IC);
    int r  = i - t * OC * IC;
    int oc = r >> 8;
    int c  = r & 255;
    wt[i] = (_Float16)w[oc * KD + c * 9 + t];
}

// ---------------- hashing pipeline ----------------

__global__ void dc_query(const float* __restrict__ x, float* __restrict__ q) {
    __shared__ float red[256];
    int c = blockIdx.x;
    float s = 0.f;
    for (int i = threadIdx.x; i < P; i += 256) s += x[c * P + i];
    red[threadIdx.x] = s;
    __syncthreads();
    for (int off = 128; off > 0; off >>= 1) {
        if (threadIdx.x < off) red[threadIdx.x] += red[threadIdx.x + off];
        __syncthreads();
    }
    if (threadIdx.x == 0) q[c] = red[0] * (1.0f / P);
}

__global__ void dc_qaddr(const float* __restrict__ rm_q, const float* __restrict__ q,
                         int* __restrict__ qaddr) {
    __shared__ int bits[NT * HS];
    int tid = threadIdx.x;
    if (tid < NT * HS) {
        const float* r = rm_q + tid * IC;
        float s = 0.f;
        for (int d = 0; d < IC; ++d) s += r[d] * q[d];
        bits[tid] = (s > 0.f) ? 1 : 0;   // relu(sign(x)): 1 iff x > 0
    }
    __syncthreads();
    if (tid < NT) {
        int a = 0;
        for (int h = 0; h < HS; ++h) a |= bits[tid * HS + h] << (HS - 1 - h);
        qaddr[tid] = a;
    }
}

__global__ void dc_waddr(const float* __restrict__ rm_w, const float* __restrict__ w,
                         int* __restrict__ waddr) {
    __shared__ float row[KD];
    __shared__ int bits[NT * HS];
    int o = blockIdx.x;
    for (int d = threadIdx.x; d < KD; d += 256) row[d] = w[o * KD + d];
    __syncthreads();
    if (threadIdx.x < NT * HS) {
        const float* r = rm_w + threadIdx.x * KD;
        float s = 0.f;
        for (int d = 0; d < KD; ++d) s += r[d] * row[d];
        bits[threadIdx.x] = (s > 0.f) ? 1 : 0;
    }
    __syncthreads();
    if (threadIdx.x < NT) {
        int a = 0;
        for (int h = 0; h < HS; ++h) a |= bits[threadIdx.x * HS + h] << (HS - 1 - h);
        waddr[threadIdx.x * OC + o] = a;
    }
}

// hist + stable top-k via rank (ties broken by lower index, matching jax.lax.top_k)
__global__ void dc_active(const int* __restrict__ waddr, const int* __restrict__ qaddr,
                          int* __restrict__ active) {
    __shared__ int hist[OC];
    int o = threadIdx.x;
    int h = 0;
    for (int t = 0; t < NT; ++t) h += (waddr[t * OC + o] == qaddr[t]);
    hist[o] = h;
    __syncthreads();
    int rank = 0;
    for (int o2 = 0; o2 < OC; ++o2) {
        int h2 = hist[o2];
        rank += (h2 > h) || (h2 == h && o2 < o);
    }
    active[o] = (h > 0 && rank < LIM) ? 1 : 0;
}

// ---------------- implicit-GEMM conv with WMMA ----------------
// block: 256 threads = 8 waves; tile 128 oc x 32 pos; wave = 16 oc x 32 pos (2 accums)
__global__ __launch_bounds__(256)
void dc_conv(const _Float16* __restrict__ xp, const _Float16* __restrict__ wt,
             float* __restrict__ y) {
    __shared__ __align__(16) _Float16 Aw[128 * 32];   // [oc][c]   8 KB
    __shared__ __align__(16) _Float16 Bx[32 * 40];    // [pos][c]  2.5 KB (stride 40 pad)

    const int oc0  = blockIdx.x * 128;
    const int pos0 = blockIdx.y * 32;
    const int tid  = threadIdx.x;
    const int lane = tid & 31;
    const int wv   = tid >> 5;

    v8f acc0 = {};
    v8f acc1 = {};

    // --- A-stage assignment: thread copies 16 halves (32 B) via 2 async b128 ---
    const int aoi  = tid >> 1;
    const int aseg = (tid & 1) * 16;
    const unsigned ldsA =
        (unsigned)(unsigned long long)(&Aw[aoi * 32 + aseg]);   // low 32 bits = LDS offset
    const long long agbase = (long long)((oc0 + aoi) * IC + aseg) * 2; // bytes into wt tap-plane

    // --- B-stage assignment: thread owns 1 position x 4 channels ---
    const int bpj = tid & 31;
    const int bcg = (tid >> 5) * 4;
    {
        // precompute nothing heavy in-loop: position base in padded plane
    }
    const int bp   = pos0 + bpj;
    const int bh   = bp / HW;
    const int bwc  = bp - bh * HW;
    const int bbase = bh * PW + bwc;          // tap (0,0) offset in padded plane
    _Float16* bdst = &Bx[bpj * 40 + bcg];

    // fragment LDS offsets per documented CDNA5 WMMA layouts
    const int am    = lane & 15;
    const int abase = (lane < 16) ? 0 : 8;             // A: K runs {abase..+7, abase+16..+23}
    const int aoff  = (wv * 16 + am) * 32 + abase;
    const int bcol  = lane & 15;
    const int kb    = (lane < 16) ? 0 : 8;             // B: K runs {kb..+7, kb+16..+23}
    const int b0off = bcol * 40 + kb;
    const int b1off = (bcol + 16) * 40 + kb;

    for (int kh = 0; kh < 3; ++kh) {
        for (int kw = 0; kw < 3; ++kw) {
            const int toff = bbase + kh * PW + kw;               // shifted gather base
            const _Float16* wsrc = wt + (kh * 3 + kw) * (OC * IC);
            for (int c0 = 0; c0 < IC; c0 += 32) {
                __syncthreads();
                // stage A: 128 oc x 32 c -> LDS via async DMA (no VGPR round-trip)
                {
                    unsigned long long ga =
                        (unsigned long long)wsrc + (unsigned long long)(agbase + (long long)c0 * 2);
                    asm volatile(
                        "global_load_async_to_lds_b128 %0, %1, off\n\t"
                        "global_load_async_to_lds_b128 %0, %1, off offset:16"
                        :: "v"(ldsA), "v"(ga) : "memory");
                }
                // stage B: 32 c x 32 pos, halo-padded gather (no bounds checks), pos-major
                {
                    const _Float16* xs = xp + (c0 + bcg) * PP + toff;
                    union { _Float16 h[4]; unsigned u[2]; } pk;
                    pk.h[0] = xs[0 * PP];
                    pk.h[1] = xs[1 * PP];
                    pk.h[2] = xs[2 * PP];
                    pk.h[3] = xs[3 * PP];
                    *(uint2*)bdst = *(const uint2*)pk.u;          // one ds_store_b64
                }
#if __has_builtin(__builtin_amdgcn_s_wait_asynccnt)
                __builtin_amdgcn_s_wait_asynccnt(0);
#else
                asm volatile("s_wait_asynccnt 0x0" ::: "memory");
#endif
                __syncthreads();

                v8h alo = *(const v8h*)(&Aw[aoff]);
                v8h ahi = *(const v8h*)(&Aw[aoff + 16]);
                v16h a = __builtin_shufflevector(alo, ahi,
                         0,1,2,3,4,5,6,7,8,9,10,11,12,13,14,15);

                v8h b0lo = *(const v8h*)(&Bx[b0off]);
                v8h b0hi = *(const v8h*)(&Bx[b0off + 16]);
                v16h b0 = __builtin_shufflevector(b0lo, b0hi,
                          0,1,2,3,4,5,6,7,8,9,10,11,12,13,14,15);

                v8h b1lo = *(const v8h*)(&Bx[b1off]);
                v8h b1hi = *(const v8h*)(&Bx[b1off + 16]);
                v16h b1 = __builtin_shufflevector(b1lo, b1hi,
                          0,1,2,3,4,5,6,7,8,9,10,11,12,13,14,15);

                acc0 = __builtin_amdgcn_wmma_f32_16x16x32_f16(false, a, false, b0,
                                                              (short)0, acc0, false, false);
                acc1 = __builtin_amdgcn_wmma_f32_16x16x32_f16(false, a, false, b1,
                                                              (short)0, acc1, false, false);
            }
        }
    }

    // epilogue: C/D layout — lane&15 = col, reg r = row (+8 for lanes 16..31)
    const int mbase = (lane < 16) ? 0 : 8;
    const int pj = lane & 15;
    for (int r = 0; r < 8; ++r) {
        int oc = oc0 + wv * 16 + mbase + r;
        y[oc * P + pos0 + pj]      = acc0[r];
        y[oc * P + pos0 + pj + 16] = acc1[r];
    }
}

// ---------------- fused BatchNorm(train) + mask + ReLU ----------------

__global__ void dc_bn(const float* __restrict__ y, const int* __restrict__ active,
                      const float* __restrict__ gamma, const float* __restrict__ beta,
                      float* __restrict__ out) {
    __shared__ float buf[P];
    __shared__ float r1[256];
    __shared__ float r2[256];
    int oc = blockIdx.x;
    float s = 0.f, s2 = 0.f;
    for (int i = threadIdx.x; i < P; i += 256) {
        float v = y[oc * P + i];
        buf[i] = v;
        s += v; s2 += v * v;
    }
    r1[threadIdx.x] = s; r2[threadIdx.x] = s2;
    __syncthreads();
    for (int off = 128; off > 0; off >>= 1) {
        if (threadIdx.x < off) {
            r1[threadIdx.x] += r1[threadIdx.x + off];
            r2[threadIdx.x] += r2[threadIdx.x + off];
        }
        __syncthreads();
    }
    float mean = r1[0] * (1.0f / P);
    float var  = r2[0] * (1.0f / P) - mean * mean;
    float a, b;
    if (active[oc]) { a = gamma[oc] * rsqrtf(var + EPS); b = beta[oc] - mean * a; }
    else            { a = 0.f;                            b = beta[oc]; }   // masked channel: relu(beta)
    for (int i = threadIdx.x; i < P; i += 256) {
        float v = a * buf[i] + b;
        out[oc * P + i] = fmaxf(v, 0.f);
    }
}

// ---------------- launch ----------------

extern "C" void kernel_launch(void* const* d_in, const int* in_sizes, int n_in,
                              void* d_out, int out_size, void* d_ws, size_t ws_size,
                              hipStream_t stream) {
    (void)in_sizes; (void)n_in; (void)out_size; (void)ws_size;
    const float* x     = (const float*)d_in[0];
    const float* w     = (const float*)d_in[1];
    const float* rm_w  = (const float*)d_in[2];
    const float* rm_q  = (const float*)d_in[3];
    const float* gamma = (const float*)d_in[4];
    const float* beta  = (const float*)d_in[5];
    float* out = (float*)d_out;

    char* ws = (char*)d_ws;
    size_t off = 0;
    auto alloc = [&](size_t bytes) -> char* {
        char* p = ws + off;
        off += (bytes + 255) & ~size_t(255);
        return p;
    };
    _Float16* xpad = (_Float16*)alloc(sizeof(_Float16) * IC * PP);      // 1.7 MB padded x
    _Float16* wt   = (_Float16*)alloc(sizeof(_Float16) * 9 * OC * IC);  // 2.4 MB
    float*    y    = (float*)   alloc(sizeof(float) * OC * P);          // 6.4 MB
    float*    q    = (float*)   alloc(sizeof(float) * IC);
    int*      qadr = (int*)     alloc(sizeof(int) * NT);
    int*      wadr = (int*)     alloc(sizeof(int) * NT * OC);
    int*      act  = (int*)     alloc(sizeof(int) * OC);

    dc_pad_x <<<(IC * PP + 255) / 256, 256, 0, stream>>>(x, xpad);
    dc_cvt_w <<<(9 * OC * IC + 255) / 256, 256, 0, stream>>>(w, wt);
    dc_query <<<IC, 256, 0, stream>>>(x, q);
    dc_qaddr <<<1, 128, 0, stream>>>(rm_q, q, qadr);
    dc_waddr <<<OC, 256, 0, stream>>>(rm_w, w, wadr);
    dc_active<<<1, OC, 0, stream>>>(wadr, qadr, act);

    dim3 grid(OC / 128, P / 32);   // (4, 98)
    dc_conv  <<<grid, 256, 0, stream>>>(xpad, wt, y);
    dc_bn    <<<OC, 256, 0, stream>>>(y, act, gamma, beta, out);
}
